// Net_50783693308231
// MI455X (gfx1250) — matrix operand order repaired and checked by reading, not compiled, not run on hardware
//
#include <hip/hip_runtime.h>

// ---------------- problem constants ----------------
#define NNODES  12
#define NEDGES  30          // also edge_attr feature count
#define FEAT    22
#define HID     192
#define OUTC    32
#define BATCH   8192
#define NTOT    (BATCH * NNODES)     // 98304
#define ETOT    (BATCH * NEDGES)     // 245760
#define OUTSP   132

// ---------------- async-to-LDS availability ----------------
#if defined(__has_builtin)
#if __has_builtin(__builtin_amdgcn_global_load_async_to_lds_b128) && \
    __has_builtin(__builtin_amdgcn_s_wait_asynccnt)
#define HAVE_ASYNC 1
#endif
#endif
#ifndef HAVE_ASYNC
#define HAVE_ASYNC 0
#endif

#if HAVE_ASYNC
// builtin prototype (from hipcc diagnostic):
//   param0: int __attribute__((vector_size(16))) __device__(AS1) *
//   param1: AS3 counterpart; params 2,3: imm offset / imm cpol
typedef int v4i_gcc __attribute__((vector_size(16)));
typedef __attribute__((address_space(1))) v4i_gcc as1_v4i;
typedef __attribute__((address_space(3))) v4i_gcc as3_v4i;
#endif

// ---------------- WMMA types ----------------
typedef __bf16 bf16x16 __attribute__((ext_vector_type(16)));
typedef float  f32x8   __attribute__((ext_vector_type(8)));

union ABu { bf16x16 v; unsigned int u[8]; };

__device__ __forceinline__ unsigned short f2bf(float f) {
    unsigned int u = __float_as_uint(f);
    u += 0x7FFFu + ((u >> 16) & 1u);        // round-to-nearest-even
    return (unsigned short)(u >> 16);
}

// =====================================================================
// Prep: convert weights to bf16, transposed [N][K] for packed B loads
// =====================================================================
__global__ __launch_bounds__(256) void prep_kernel(
    const float* __restrict__ w2, const float* __restrict__ l1,
    const float* __restrict__ l2, const float* __restrict__ l3,
    unsigned short* __restrict__ w2bT, unsigned short* __restrict__ l1bT,
    unsigned short* __restrict__ l2bT, unsigned short* __restrict__ l3bT)
{
    int i = blockIdx.x * 256 + threadIdx.x;
    if (i < 704 * 192) { int n = i / 192, k = i % 192; w2bT[i] = f2bf(w2[k * 704 + n]); return; }
    i -= 704 * 192;
    if (i < 96 * 384)  { int n = i / 384, k = i % 384; l1bT[i] = f2bf(l1[k * 96 + n]); return; }
    i -= 96 * 384;
    if (i < 48 * 96)   { int n = i / 96,  k = i % 96;  l2bT[i] = f2bf(l2[k * 48 + n]); return; }
    i -= 48 * 96;
    if (i < 144 * 64)  { int n = i / 64,  k = i % 64;
                         l3bT[i] = (n < OUTSP && k < 48) ? f2bf(l3[k * OUTSP + n]) : (unsigned short)0; }
}

// =====================================================================
// Fused edge kernel: 4 waves/block, 16 edges/wave, 64 edges/block.
//   stage A: h = relu(edge_attr @ w1 + b1)            (WMMA, K pad 30->32)
//   stage B: per channel c, slice of w2 staged in LDS (async, dbl-buffered,
//            shared by all 4 waves); Wc = h @ slice (WMMA, K=192);
//            msg[:,o] += x[src][:,c] * (Wc + b2_c)
//   atomicAdd msg into agg[dst]
// =====================================================================
__global__ __launch_bounds__(128) void edge_kernel(
    const float* __restrict__ x, const float* __restrict__ ea,
    const int*   __restrict__ eidx,
    const float* __restrict__ w1, const float* __restrict__ b1,
    const float* __restrict__ b2,
    const unsigned short* __restrict__ w2bT,
    float* __restrict__ agg)
{
    // s_wbuf: stage A = w1T [192n][32k] (12288B);
    //         stage B = two 12288B slice buffers of w2bT [32col][192k]
    __shared__ __align__(16) unsigned short s_wbuf[2 * 6144];
    __shared__ float s_b1[HID];
    __shared__ float s_b2[FEAT * OUTC];
    __shared__ __align__(16) unsigned short s_h[4][16 * HID];  // per-wave h (bf16)
    __shared__ float s_y[4][16 * 24];                          // per-wave x[src]

    const int tid = threadIdx.x;
    for (int i = tid; i < HID * 32; i += 128) {
        int n = i >> 5, k = i & 31;
        s_wbuf[i] = (k < NEDGES) ? f2bf(w1[k * HID + n]) : (unsigned short)0;
    }
    for (int i = tid; i < HID; i += 128)         s_b1[i] = b1[i];
    for (int i = tid; i < FEAT * OUTC; i += 128) s_b2[i] = b2[i];
    __syncthreads();

    const int w   = tid >> 5;
    const int lid = tid & 31;
    const int llo = lid & 15;
    const int lhi = lid >> 4;
    const int e0  = blockIdx.x * 64 + w * 16;

    // gather source-node features (f32; used as per-edge scalars)
    for (int i = lid; i < 16 * FEAT; i += 32) {
        int m = i / FEAT, c = i % FEAT;
        int s = eidx[e0 + m];
        s_y[w][m * 24 + c] = x[s * FEAT + c];
    }

    // A fragment: 16 edge_attr rows, K padded 30 -> 32
    ABu afrag;
    {
        const int row = e0 + llo;
        #pragma unroll
        for (int j = 0; j < 8; ++j) {
            const int k0 = 2 * (j & 3) + 16 * (j >> 2) + 8 * lhi;
            float v0 = (k0     < NEDGES) ? ea[row * NEDGES + k0]     : 0.f;
            float v1 = (k0 + 1 < NEDGES) ? ea[row * NEDGES + k0 + 1] : 0.f;
            afrag.u[j] = (unsigned int)f2bf(v0) | ((unsigned int)f2bf(v1) << 16);
        }
    }

    // ---- stage A: h = relu(ea @ w1 + b1), h -> LDS as bf16 [16][192] ----
    for (int nt = 0; nt < 12; ++nt) {
        ABu bfrag;
        #pragma unroll
        for (int j = 0; j < 8; ++j) {
            const int k = 2 * (j & 3) + 16 * (j >> 2) + 8 * lhi;
            bfrag.u[j] = *(const unsigned int*)&s_wbuf[(nt * 16 + llo) * 32 + k];
        }
        f32x8 acc = {};
        acc = __builtin_amdgcn_wmma_f32_16x16x32_bf16(false, afrag.v, false, bfrag.v,
                                                      (short)0, acc, false, false);
        #pragma unroll
        for (int r = 0; r < 8; ++r) {
            const int m = r + 8 * lhi;
            const int col = nt * 16 + llo;
            float v = acc[r] + s_b1[col];
            v = v > 0.f ? v : 0.f;
            s_h[w][m * HID + col] = f2bf(v);
        }
    }

    // hoist stage-B A-fragments (h) into registers: invariant across c
    ABu ah[6];
    #pragma unroll
    for (int ks = 0; ks < 6; ++ks) {
        #pragma unroll
        for (int j = 0; j < 8; ++j) {
            const int k0 = 2 * (j & 3) + 16 * (j >> 2) + 8 * lhi + 32 * ks;
            ah[ks].u[j] = *(const unsigned int*)&s_h[w][llo * HID + k0];
        }
    }

    __syncthreads();   // all waves done with w1T region of s_wbuf

    // cooperative slice stage: 12288B = 6 x b128 per thread (128 threads)
    auto load_slice = [&](int cslice, int buf) {
        const char* g = (const char*)(w2bT + (size_t)cslice * 32 * HID);
        char* l = (char*)&s_wbuf[buf * 6144];
#if HAVE_ASYNC
        #pragma unroll
        for (int i = 0; i < 6; ++i)
            __builtin_amdgcn_global_load_async_to_lds_b128(
                (as1_v4i*)(g + tid * 16 + i * 2048),
                (as3_v4i*)(l + tid * 16 + i * 2048), 0, 0);
#else
        #pragma unroll
        for (int i = 0; i < 6; ++i) {
            uint4 v = *(const uint4*)(g + tid * 16 + i * 2048);
            *(uint4*)(l + tid * 16 + i * 2048) = v;
        }
#endif
    };

    load_slice(0, 0);

    // ---- stage B ----
    f32x8 macc[2] = {};
    for (int c = 0; c < FEAT; ++c) {
        const int cur = c & 1;
        if (c) __syncthreads();                 // everyone done reading buf cur^1
        if (c + 1 < FEAT) {
            load_slice(c + 1, cur ^ 1);
#if HAVE_ASYNC
            __builtin_amdgcn_s_wait_asynccnt(6);   // newest 6 in flight => slice c done
#endif
        } else {
#if HAVE_ASYNC
            __builtin_amdgcn_s_wait_asynccnt(0);
#endif
        }
        __syncthreads();                        // slice c visible to all waves

        #pragma unroll
        for (int nt = 0; nt < 2; ++nt) {
            f32x8 acc = {};
            #pragma unroll
            for (int ks = 0; ks < 6; ++ks) {
                ABu bfrag;
                #pragma unroll
                for (int j = 0; j < 8; ++j) {
                    const int kb = 2 * (j & 3) + 16 * (j >> 2) + 8 * lhi + 32 * ks;
                    bfrag.u[j] = *(const unsigned int*)
                        &s_wbuf[cur * 6144 + (nt * 16 + llo) * HID + kb];
                }
                acc = __builtin_amdgcn_wmma_f32_16x16x32_bf16(false, ah[ks].v, false, bfrag.v,
                                                              (short)0, acc, false, false);
            }
            const int ncol = c * OUTC + nt * 16 + llo;
            #pragma unroll
            for (int r = 0; r < 8; ++r) {
                const int m = r + 8 * lhi;
                macc[nt][r] += s_y[w][m * 24 + c] * (acc[r] + s_b2[ncol]);
            }
        }
    }

    // ---- scatter-add into agg[dst] ----
    #pragma unroll
    for (int r = 0; r < 8; ++r) {
        const int m = r + 8 * lhi;
        const int d = eidx[ETOT + e0 + m];
        atomicAdd(&agg[d * OUTC + llo],      macc[0][r]);
        atomicAdd(&agg[d * OUTC + 16 + llo], macc[1][r]);
    }
}

// =====================================================================
// Node kernel: g = relu(x @ root1 + agg + bias1) -> bf16
// =====================================================================
__global__ __launch_bounds__(256) void node_kernel(
    const float* __restrict__ x, const float* __restrict__ root1,
    const float* __restrict__ bias1, const float* __restrict__ agg,
    unsigned short* __restrict__ gb)
{
    __shared__ float s_root[FEAT * OUTC];
    __shared__ float s_bias[OUTC];
    const int tid = threadIdx.x;
    for (int i = tid; i < FEAT * OUTC; i += 256) s_root[i] = root1[i];
    if (tid < OUTC) s_bias[tid] = bias1[tid];
    __syncthreads();

    const int gid = blockIdx.x * 256 + tid;
    const int n = gid >> 5, o = gid & 31;
    float acc = agg[gid] + s_bias[o];
    #pragma unroll
    for (int c = 0; c < FEAT; ++c)
        acc += x[n * FEAT + c] * s_root[c * OUTC + o];
    acc = acc > 0.f ? acc : 0.f;
    gb[gid] = f2bf(acc);
}

// =====================================================================
// Head GEMM: 16 rows per wave, bf16 WMMA, relu.
//   MODE 1: bf16 out   MODE 2: bf16 out + zero pad cols 48..63
//   MODE 3: f32 out, cols < 132 only
// =====================================================================
template <int MODE>
__global__ __launch_bounds__(256) void head_kernel(
    const unsigned short* __restrict__ A,
    const unsigned short* __restrict__ Bw,
    int K, int ntiles,
    const float* __restrict__ bias,
    unsigned short* __restrict__ outb, int ldo,
    float* __restrict__ outf)
{
    const int tid = threadIdx.x;
    const int wv  = tid >> 5;
    const int lid = tid & 31;
    const int llo = lid & 15, lhi = lid >> 4;
    const int row0 = blockIdx.x * 128 + wv * 16;
    const int ksteps = K >> 5;
    const unsigned short* ap = A + (row0 + llo) * K;

    for (int nt = 0; nt < ntiles; ++nt) {
        f32x8 acc = {};
        const unsigned short* bp = Bw + (nt * 16 + llo) * K;
        for (int ks = 0; ks < ksteps; ++ks) {
            ABu a, b;
            #pragma unroll
            for (int j = 0; j < 8; ++j) {
                const int k0 = 2 * (j & 3) + 16 * (j >> 2) + 8 * lhi + 32 * ks;
                a.u[j] = *(const unsigned int*)&ap[k0];
                const int kb = 2 * (j & 3) + 16 * (j >> 2) + 8 * lhi + 32 * ks;
                b.u[j] = *(const unsigned int*)&bp[kb];
            }
            acc = __builtin_amdgcn_wmma_f32_16x16x32_bf16(false, a.v, false, b.v,
                                                          (short)0, acc, false, false);
        }
        #pragma unroll
        for (int r = 0; r < 8; ++r) {
            const int m = row0 + r + 8 * lhi;
            const int col = nt * 16 + llo;
            if (MODE == 3) {
                if (col < OUTSP) {
                    float v = acc[r] + bias[col];
                    outf[m * OUTSP + col] = v > 0.f ? v : 0.f;
                }
            } else {
                float v = acc[r] + bias[col];
                v = v > 0.f ? v : 0.f;
                outb[m * ldo + col] = f2bf(v);
            }
        }
    }
    if (MODE == 2) {   // zero K-padding cols 48..63 for the next layer
        #pragma unroll
        for (int r = 0; r < 8; ++r) {
            const int m = row0 + r + 8 * lhi;
            outb[m * ldo + 48 + llo] = 0;
        }
    }
}

// =====================================================================
extern "C" void kernel_launch(void* const* d_in, const int* in_sizes, int n_in,
                              void* d_out, int out_size, void* d_ws, size_t ws_size,
                              hipStream_t stream)
{
    (void)in_sizes; (void)n_in; (void)out_size; (void)ws_size;
    const float* x     = (const float*)d_in[0];
    const float* ea    = (const float*)d_in[1];
    const int*   eidx  = (const int*)d_in[2];
    const float* w1    = (const float*)d_in[3];
    const float* b1    = (const float*)d_in[4];
    const float* w2    = (const float*)d_in[5];
    const float* b2    = (const float*)d_in[6];
    const float* root1 = (const float*)d_in[7];
    const float* bias1 = (const float*)d_in[8];
    const float* l1w   = (const float*)d_in[9];
    const float* l1b   = (const float*)d_in[10];
    const float* l2w   = (const float*)d_in[11];
    const float* l2b   = (const float*)d_in[12];
    const float* l3w   = (const float*)d_in[13];
    const float* l3b   = (const float*)d_in[14];
    float* out = (float*)d_out;

    char* ws = (char*)d_ws;
    size_t off = 0;
    auto alloc = [&](size_t bytes) -> void* {
        void* p = ws + off;
        off += (bytes + 255) & ~(size_t)255;
        return p;
    };
    float*          agg  = (float*)         alloc((size_t)NTOT * OUTC * 4);
    unsigned short* gb   = (unsigned short*)alloc((size_t)NTOT * OUTC * 2);
    unsigned short* h1b  = (unsigned short*)alloc((size_t)BATCH * 96 * 2);
    unsigned short* h2b  = (unsigned short*)alloc((size_t)BATCH * 64 * 2);
    unsigned short* w2bT = (unsigned short*)alloc((size_t)704 * 192 * 2);
    unsigned short* l1bT = (unsigned short*)alloc((size_t)96 * 384 * 2);
    unsigned short* l2bT = (unsigned short*)alloc((size_t)48 * 96 * 2);
    unsigned short* l3bT = (unsigned short*)alloc((size_t)144 * 64 * 2);

    (void)hipMemsetAsync(agg, 0, (size_t)NTOT * OUTC * 4, stream);

    // 135168 + 36864 + 4608 + 9216 = 185856 = 726 * 256
    prep_kernel<<<726, 256, 0, stream>>>(w2, l1w, l2w, l3w, w2bT, l1bT, l2bT, l3bT);

    edge_kernel<<<ETOT / 64, 128, 0, stream>>>(x, ea, eidx, w1, b1, b2, w2bT, agg);

    node_kernel<<<(NTOT * OUTC) / 256, 256, 0, stream>>>(x, root1, bias1, agg, gb);

    head_kernel<1><<<BATCH / 128, 256, 0, stream>>>(gb,  l1bT, 384, 6, l1b, h1b, 96, nullptr);
    head_kernel<2><<<BATCH / 128, 256, 0, stream>>>(h1b, l2bT,  96, 3, l2b, h2b, 64, nullptr);
    head_kernel<3><<<BATCH / 128, 256, 0, stream>>>(h2b, l3bT,  64, 9, l3b, nullptr, 0, out);
}
